// Quantization_76630806495966
// MI455X (gfx1250) — compile-verified, matrix operand order checked by reading, not run on hardware
//
#include <hip/hip_runtime.h>
#include <hip/hip_bf16.h>

typedef __attribute__((ext_vector_type(16))) __bf16 vbf16;
typedef __attribute__((ext_vector_type(8)))  float v8f;

#define N_TOKENS  32768
#define LATENT    512
#define N_EMBED   8192
#define BM        128
#define BN        128
#define BK        32
#define LDS_STRIDE 40   // bf16 elems/row: 32 data + 8 pad = 80B (16B aligned, conflict-free)

#if defined(__HIP_DEVICE_COMPILE__) && __has_builtin(__builtin_amdgcn_global_load_async_to_lds_b128)
#define HAVE_ASYNC_LDS 1
#else
#define HAVE_ASYNC_LDS 0
#endif

#if HAVE_ASYNC_LDS
typedef int vint4_t __attribute__((vector_size(16)));
typedef __attribute__((address_space(1))) vint4_t* gvi4_ptr;   // global int4*
typedef __attribute__((address_space(3))) vint4_t* lvi4_ptr;   // LDS int4*
#endif

__device__ __forceinline__ void wait_async_all() {
#if HAVE_ASYNC_LDS
#if __has_builtin(__builtin_amdgcn_s_wait_asynccnt)
  __builtin_amdgcn_s_wait_asynccnt(0);
#else
  asm volatile("s_wait_asynccnt 0x0" ::: "memory");
#endif
#endif
}

__device__ __forceinline__ unsigned short f2bf(float f) {
  union { float f; unsigned u; } v; v.f = f;
  unsigned u = v.u + 0x7fffu + ((v.u >> 16) & 1u);   // round-to-nearest-even
  return (unsigned short)(u >> 16);
}

__device__ __forceinline__ uint4 pack8(float4 a, float4 b) {
  uint4 r;
  r.x = (unsigned)f2bf(a.x) | ((unsigned)f2bf(a.y) << 16);
  r.y = (unsigned)f2bf(a.z) | ((unsigned)f2bf(a.w) << 16);
  r.z = (unsigned)f2bf(b.x) | ((unsigned)f2bf(b.y) << 16);
  r.w = (unsigned)f2bf(b.z) | ((unsigned)f2bf(b.w) << 16);
  return r;
}

// ---------------- one-shot f32 -> bf16 conversion (bandwidth-bound) ----------------
__global__ __launch_bounds__(256)
void cvt_bf16_kernel(const float* __restrict__ in, uint4* __restrict__ out) {
  const size_t i = (size_t)blockIdx.x * 256 + threadIdx.x;   // one uint4 = 8 bf16 per thread
  const float4* s = (const float4*)in + i * 2;
  out[i] = pack8(s[0], s[1]);
}

// ---------------- codebook row norms + zero loss slot ----------------
__global__ __launch_bounds__(256)
void vq_wn2_kernel(const float* __restrict__ W, float* __restrict__ wn2,
                   float* __restrict__ loss_out) {
  if (blockIdx.x == 0 && threadIdx.x == 0) *loss_out = 0.0f;
  const int code = blockIdx.x * 8 + (threadIdx.x >> 5);
  const int lane = threadIdx.x & 31;
  const float4* row = (const float4*)(W + (size_t)code * LATENT);
  float s = 0.0f;
#pragma unroll
  for (int j = 0; j < 4; ++j) {
    float4 v = row[lane + j * 32];
    s += v.x * v.x + v.y * v.y + v.z * v.z + v.w * v.w;
  }
#pragma unroll
  for (int off = 1; off < 32; off <<= 1) s += __shfl_xor(s, off, 32);
  if (lane == 0) wn2[code] = s;
}

// Stage one row-half (16 bf16 = 32B) of a 128x32 tile into LDS.
// PRECONV: bf16 source -> async DMA to LDS (or plain copy). Else: f32 source, convert inline.
template<bool PRECONV>
__device__ __forceinline__ void stage_half(const void* base, int row, int half,
                                           int k0, unsigned short* tile) {
  if (PRECONV) {
    const unsigned short* g = (const unsigned short*)base + (size_t)row * LATENT + k0 + half * 16;
    unsigned short* l = &tile[row * LDS_STRIDE + half * 16];
#if HAVE_ASYNC_LDS
    __builtin_amdgcn_global_load_async_to_lds_b128(
        (gvi4_ptr)(void*)g, (lvi4_ptr)(void*)l, 0, 0);
    __builtin_amdgcn_global_load_async_to_lds_b128(
        (gvi4_ptr)(void*)g, (lvi4_ptr)(void*)l, 16, 0);
#else
    const uint4* s = (const uint4*)g;
    uint4* dst = (uint4*)l;
    dst[0] = s[0];
    dst[1] = s[1];
    if (k0 + BK < LATENT) __builtin_prefetch(g + BK, 0, 3);
#endif
  } else {
    const float4* s = (const float4*)((const float*)base + (size_t)row * LATENT + k0 + half * 16);
    uint4* dst = (uint4*)&tile[row * LDS_STRIDE + half * 16];
    dst[0] = pack8(s[0], s[1]);
    dst[1] = pack8(s[2], s[3]);
  }
}

// ---------------- WMMA distance GEMM + running argmin ----------------
// 256 threads = 8 waves in a 4(M) x 2(N) grid; each wave owns 32x64 of the 128x128 tile.
template<bool PRECONV>
__global__ __launch_bounds__(256)
void vq_argmin_kernel(const void* __restrict__ xin, const void* __restrict__ win,
                      const float* __restrict__ wn2, int* __restrict__ ids_ws,
                      float* __restrict__ ids_out) {
  __shared__ __align__(16) unsigned short As[2][BM * LDS_STRIDE];
  __shared__ __align__(16) unsigned short Bs[2][BN * LDS_STRIDE];

  const int tid  = threadIdx.x;
  const int wave = tid >> 5;
  const int lane = tid & 31;
  const int m0   = blockIdx.x * BM;

  const int ldRow  = tid >> 1;      // 0..127 : tile row staged by this thread
  const int ldHalf = tid & 1;       // 0/1    : which 16-col half

  const int mw = (wave & 3) * 32;   // wave origin: 4 waves along M
  const int nw = (wave >> 2) * 64;  //             2 waves along N

  const int lmod  = lane & 15;
  const int khalf = lane >> 4;      // K-half select per ISA 16-bit A/B layout

  const char* xbase = (const char*)xin +
      (PRECONV ? (size_t)m0 * LATENT * 2 : (size_t)m0 * LATENT * 4);

  float best[2][8];
  int   bestid[2][8];
#pragma unroll
  for (int mt = 0; mt < 2; ++mt)
#pragma unroll
    for (int r = 0; r < 8; ++r) { best[mt][r] = 3.402823466e38f; bestid[mt][r] = 0; }

  for (int n0 = 0; n0 < N_EMBED; n0 += BN) {
    const char* wbase = (const char*)win +
        (PRECONV ? (size_t)n0 * LATENT * 2 : (size_t)n0 * LATENT * 4);

    v8f acc[2][4];
#pragma unroll
    for (int mt = 0; mt < 2; ++mt)
#pragma unroll
      for (int nt = 0; nt < 4; ++nt)
#pragma unroll
        for (int e = 0; e < 8; ++e) acc[mt][nt][e] = 0.0f;

    // prologue: stage k0 = 0 into buffer 0
    stage_half<PRECONV>(xbase, ldRow, ldHalf, 0, As[0]);
    stage_half<PRECONV>(wbase, ldRow, ldHalf, 0, Bs[0]);
    wait_async_all();
    __syncthreads();

    int buf = 0;
    for (int k0 = 0; k0 < LATENT; k0 += BK) {
      const int nxt = buf ^ 1;
      if (k0 + BK < LATENT) {          // stage next slab while computing this one
        stage_half<PRECONV>(xbase, ldRow, ldHalf, k0 + BK, As[nxt]);
        stage_half<PRECONV>(wbase, ldRow, ldHalf, k0 + BK, Bs[nxt]);
      }

      union Frag { vbf16 v; uint4 q[2]; };
      Frag a[2], b[4];
#pragma unroll
      for (int mt = 0; mt < 2; ++mt) {
        const unsigned short* p = &As[buf][(mw + mt * 16 + lmod) * LDS_STRIDE];
        a[mt].q[0] = *(const uint4*)(p + khalf * 8);        // K = khalf*8 .. +7
        a[mt].q[1] = *(const uint4*)(p + 16 + khalf * 8);   // K = 16+khalf*8 .. +7
      }
#pragma unroll
      for (int nt = 0; nt < 4; ++nt) {
        const unsigned short* p = &Bs[buf][(nw + nt * 16 + lmod) * LDS_STRIDE];
        b[nt].q[0] = *(const uint4*)(p + khalf * 8);
        b[nt].q[1] = *(const uint4*)(p + 16 + khalf * 8);
      }
#pragma unroll
      for (int mt = 0; mt < 2; ++mt)
#pragma unroll
        for (int nt = 0; nt < 4; ++nt)
          acc[mt][nt] = __builtin_amdgcn_wmma_f32_16x16x32_bf16(
              false, a[mt].v, false, b[nt].v,
              (short)0, acc[mt][nt], false, false);

      wait_async_all();                // next-slab DMA into LDS complete
      __syncthreads();
      buf = nxt;
    }

    // fold this 128-wide code slab into the running per-lane argmin
#pragma unroll
    for (int nt = 0; nt < 4; ++nt) {
      const int col = n0 + nw + nt * 16 + lmod;
      const float w2 = wn2[col];
#pragma unroll
      for (int mt = 0; mt < 2; ++mt)
#pragma unroll
        for (int r = 0; r < 8; ++r) {
          float d = fmaf(-2.0f, acc[mt][nt][r], w2);   // ||w||^2 - 2 x.w
          if (d < best[mt][r]) { best[mt][r] = d; bestid[mt][r] = col; }
        }
    }
  }

  // cross-lane argmin over the 16 columns each row is spread across
#pragma unroll
  for (int mt = 0; mt < 2; ++mt)
#pragma unroll
    for (int r = 0; r < 8; ++r) {
      float v = best[mt][r];
      int   id = bestid[mt][r];
#pragma unroll
      for (int off = 1; off < 16; off <<= 1) {
        float ov = __shfl_xor(v, off, 32);
        int   oi = __shfl_xor(id, off, 32);
        if (ov < v || (ov == v && oi < id)) { v = ov; id = oi; }
      }
      if (lmod == 0) {
        const int token = m0 + mw + mt * 16 + khalf * 8 + r;  // C-layout: VGPR r -> M=r(+8)
        ids_ws[token]  = id;
        ids_out[token] = (float)id;
      }
    }
}

// ---------------- gather emb_out + exact f32 loss ----------------
__global__ __launch_bounds__(256)
void vq_gather_kernel(const float* __restrict__ x, const float* __restrict__ W,
                      const int* __restrict__ ids, float* __restrict__ emb_out,
                      float* __restrict__ loss_out) {
  const int grp   = threadIdx.x >> 7;
  const int t     = threadIdx.x & 127;
  const int token = blockIdx.x * 2 + grp;
  const int id    = ids[token];

  const float4* wrow = (const float4*)(W + (size_t)id * LATENT);
  const float4* xrow = (const float4*)(x + (size_t)token * LATENT);
  float4*       orow = (float4*)(emb_out + (size_t)token * LATENT);

  float4 w4 = wrow[t];
  float4 x4 = xrow[t];
  orow[t] = w4;                                    // emb_out = x + sg(emb - x) == emb
  float dx = x4.x - w4.x, dy = x4.y - w4.y, dz = x4.z - w4.z, dw = x4.w - w4.w;
  float s = dx * dx + dy * dy + dz * dz + dw * dw;
#pragma unroll
  for (int off = 1; off < 32; off <<= 1) s += __shfl_xor(s, off, 32);
  if ((threadIdx.x & 31) == 0)
    atomicAdd(loss_out, s * (1.25f / (float)N_TOKENS));
}

extern "C" void kernel_launch(void* const* d_in, const int* in_sizes, int n_in,
                              void* d_out, int out_size, void* d_ws, size_t ws_size,
                              hipStream_t stream) {
  const float* x = (const float*)d_in[0];
  // d_in[1] = temperature (unused by the reference math path)
  const float* W = (const float*)d_in[2];

  float* out      = (float*)d_out;
  float* emb_out  = out;                                    // [32768*512]
  float* ids_out  = out + (size_t)N_TOKENS * LATENT;        // [32768]
  float* loss_out = ids_out + N_TOKENS;                     // [1]

  // workspace layout
  char*  ws     = (char*)d_ws;
  float* wn2    = (float*)ws;                                // 8192 f32   (32 KB)
  int*   ids_ws = (int*)(ws + 32 * 1024);                    // 32768 i32  (128 KB)
  const size_t XB_OFF = 256 * 1024;                          // 256 KB aligned
  const size_t XB_SZ  = (size_t)N_TOKENS * LATENT * 2;       // 32 MB bf16 x
  const size_t WB_OFF = XB_OFF + XB_SZ;
  const size_t WB_SZ  = (size_t)N_EMBED * LATENT * 2;        // 8 MB bf16 W
  const bool preconv  = ws_size >= WB_OFF + WB_SZ;

  vq_wn2_kernel<<<N_EMBED / 8, 256, 0, stream>>>(W, wn2, loss_out);

  if (preconv) {
    unsigned short* xb = (unsigned short*)(ws + XB_OFF);
    unsigned short* wb = (unsigned short*)(ws + WB_OFF);
    cvt_bf16_kernel<<<(N_TOKENS * (LATENT / 8)) / 256, 256, 0, stream>>>(x, (uint4*)xb);
    cvt_bf16_kernel<<<(N_EMBED  * (LATENT / 8)) / 256, 256, 0, stream>>>(W, (uint4*)wb);
    vq_argmin_kernel<true><<<N_TOKENS / BM, 256, 0, stream>>>(xb, wb, wn2, ids_ws, ids_out);
  } else {
    vq_argmin_kernel<false><<<N_TOKENS / BM, 256, 0, stream>>>(x, W, wn2, ids_ws, ids_out);
  }

  vq_gather_kernel<<<N_TOKENS / 2, 256, 0, stream>>>(x, W, ids_ws, emb_out, loss_out);
}
